// CompiledContrastiveLoss_87308095193294
// MI455X (gfx1250) — matrix operand order repaired and checked by reading, not compiled, not run on hardware
//
#include <hip/hip_runtime.h>
#include <hip/hip_bf16.h>

typedef _Float16 v16h __attribute__((ext_vector_type(16)));
typedef _Float16 v8h  __attribute__((ext_vector_type(8)));
typedef float    v8f  __attribute__((ext_vector_type(8)));

#define BATCH 8192
#define DIM   256
#define NTILE (BATCH / 16)      // 512 column tiles of 16
#define INV_SQRT_T 1.1952286093343936f   // 1/sqrt(0.7)
#define CMAX 1.4285714285714286f         // 1/0.7 = upper bound of sims/T
#define BIG_NEG (-1.0e30f)

// ---------------------------------------------------------------------------
// Kernel 0: zero the scalar accumulators (loss sum, valid count)
// ---------------------------------------------------------------------------
__global__ void init_acc_kernel(float* acc) {
    acc[0] = 0.0f;
    acc[1] = 0.0f;
}

// ---------------------------------------------------------------------------
// Kernel 1: L2-normalize rows, fold in 1/sqrt(T), convert to f16.
// One wave (32 lanes) per row, 8 consecutive floats per lane.
// ---------------------------------------------------------------------------
__global__ __launch_bounds__(256) void normalize_kernel(
    const float* __restrict__ emb, _Float16* __restrict__ e16) {
    const int lane = threadIdx.x & 31;
    const int wave = threadIdx.x >> 5;
    const int row  = blockIdx.x * 8 + wave;

    const float* src = emb + (size_t)row * DIM + lane * 8;
    float x[8];
    const float4* p4 = (const float4*)src;
    float4 a = p4[0], b = p4[1];
    x[0]=a.x; x[1]=a.y; x[2]=a.z; x[3]=a.w;
    x[4]=b.x; x[5]=b.y; x[6]=b.z; x[7]=b.w;

    float ss = 0.0f;
#pragma unroll
    for (int i = 0; i < 8; ++i) ss += x[i] * x[i];
#pragma unroll
    for (int off = 1; off < 32; off <<= 1)
        ss += __shfl_xor(ss, off, 32);

    const float scale = rsqrtf(ss) * INV_SQRT_T;

    v8h o;
#pragma unroll
    for (int i = 0; i < 8; ++i) o[i] = (_Float16)(x[i] * scale);
    *(v8h*)(e16 + (size_t)row * DIM + lane * 8) = o;
}

// ---------------------------------------------------------------------------
// Kernel 2: fused Gram-matrix (WMMA f16 -> f32) + branchless streaming
// masked reductions.  Because |sims/T| <= CMAX, the stable LSE uses the
// constant shift CMAX; no online max, no divergent updates:
//     ns += isneg ? exp(v - CMAX) : 0       (one v_exp_f32 + cndmask)
//     pm  = ispos ? max(pm, v)    : pm      (v_cndmask + v_max)
// One block per 16-anchor tile; 8 waves split the 512 column tiles.
// ---------------------------------------------------------------------------
__global__ __launch_bounds__(256) void contrastive_kernel(
    const _Float16* __restrict__ e16, const int* __restrict__ cats,
    float* __restrict__ acc) {
    __shared__ int   scat[BATCH];          // 32 KB category cache
    __shared__ float red[8][16][2];        // per-wave partial (pm, ns)

    const int tid  = threadIdx.x;
    const int wave = tid >> 5;
    const int lane = tid & 31;
    const int hi   = lane >> 4;            // 0: lanes 0-15, 1: lanes 16-31
    const int lo   = lane & 15;
    const int i0   = blockIdx.x * 16;      // anchor tile base row

    // Stage all categories into LDS.
    for (int t = tid; t < BATCH; t += 256) scat[t] = cats[t];
    __syncthreads();

    // --- A fragments (anchor tile, K = 0..255), contiguous-k A layout (same order as B):
    // lane m / m+16 both hold row M=m; lanes<16: K = k0+0..15,
    // lanes>=16: K = k0+16..31 (2 halves per VGPR).
    v16h afrag[8];
    const _Float16* arow = e16 + (size_t)(i0 + lo) * DIM;
#pragma unroll
    for (int kk = 0; kk < 8; ++kk) {
        const int kbase = kk * 32 + hi * 16;
        union { v16h v; v8h h[2]; } u;
        u.h[0] = *(const v8h*)(arow + kbase);
        u.h[1] = *(const v8h*)(arow + kbase + 8);
        afrag[kk] = u.v;
    }

    // Anchor categories / row ids for this lane's 8 C rows (M = hi*8 + r).
    int acat[8], irow[8];
#pragma unroll
    for (int r = 0; r < 8; ++r) {
        irow[r] = i0 + hi * 8 + r;
        acat[r] = scat[irow[r]];
    }

    float pm[8], ns[8];
#pragma unroll
    for (int r = 0; r < 8; ++r) { pm[r] = BIG_NEG; ns[r] = 0.0f; }

    for (int jt = wave; jt < NTILE; jt += 8) {
        const int j0   = jt * 16;
        const int jcol = j0 + lo;          // column this lane owns in C
        const int ccat = scat[jcol];

        // B fragment: column n of B-tile = row (j0+n) of e16.
        // lanes<16 hold K = k0..k0+15, lanes>=16 hold K = k0+16..k0+31.
        const _Float16* brow = e16 + (size_t)jcol * DIM;
        v8f c = {};
#pragma unroll
        for (int kk = 0; kk < 8; ++kk) {
            const v16h b = *(const v16h*)(brow + kk * 32 + hi * 16);
            c = __builtin_amdgcn_wmma_f32_16x16x32_f16(
                    false, afrag[kk], false, b, (short)0, c, false, false);
        }

        // C layout: c[r] = sims(i0 + hi*8 + r, j0 + lo) (already /T).
#pragma unroll
        for (int r = 0; r < 8; ++r) {
            const float v     = c[r];
            const bool  isneg = (ccat != acat[r]);
            const bool  ispos = !isneg && (jcol != irow[r]);
            const float ev    = __expf(v - CMAX);
            ns[r] += isneg ? ev : 0.0f;
            pm[r]  = ispos ? fmaxf(pm[r], v) : pm[r];
        }
    }

    // Combine the 16 column-slot lanes that share each row: max / sum only.
#pragma unroll
    for (int r = 0; r < 8; ++r) {
#pragma unroll
        for (int off = 1; off < 16; off <<= 1) {
            pm[r] = fmaxf(pm[r], __shfl_xor(pm[r], off, 32));
            ns[r] += __shfl_xor(ns[r], off, 32);
        }
    }

    if (lo == 0) {
#pragma unroll
        for (int r = 0; r < 8; ++r) {
            red[wave][hi * 8 + r][0] = pm[r];
            red[wave][hi * 8 + r][1] = ns[r];
        }
    }
    __syncthreads();

    // Wave 0, lanes 0-15: merge the 8 wave partials for one row each,
    // compute the per-anchor loss, accumulate globally.
    if (wave == 0 && lane < 16) {
        float P = BIG_NEG, S = 0.0f;
#pragma unroll
        for (int w = 0; w < 8; ++w) {
            P = fmaxf(P, red[w][lane][0]);
            S += red[w][lane][1];
        }
        const bool valid = (P > -5.0e29f) && (S > 0.0f);
        if (valid) {
            // lse = CMAX + log(exp(P - CMAX) + S);  loss = lse - P
            const float lse = CMAX + __logf(__expf(P - CMAX) + S);
            atomicAdd(&acc[0], lse - P);
            atomicAdd(&acc[1], 1.0f);
        }
    }
}

// ---------------------------------------------------------------------------
// Kernel 3: finalize  out = loss_sum / max(n_valid, 1)
// ---------------------------------------------------------------------------
__global__ void finalize_kernel(const float* acc, float* out) {
    out[0] = acc[0] / fmaxf(acc[1], 1.0f);
}

// ---------------------------------------------------------------------------
extern "C" void kernel_launch(void* const* d_in, const int* in_sizes, int n_in,
                              void* d_out, int out_size, void* d_ws, size_t ws_size,
                              hipStream_t stream) {
    (void)in_sizes; (void)n_in; (void)out_size; (void)ws_size;
    const float* emb  = (const float*)d_in[0];
    const int*   cats = (const int*)d_in[1];
    // d_in[2] (font_labels) unused by the loss.

    float*     acc = (float*)d_ws;                        // 2 floats
    _Float16*  e16 = (_Float16*)((char*)d_ws + 256);      // 4 MB f16 embeddings
    float*     out = (float*)d_out;

    init_acc_kernel<<<1, 1, 0, stream>>>(acc);
    normalize_kernel<<<BATCH / 8, 256, 0, stream>>>(emb, e16);
    contrastive_kernel<<<BATCH / 16, 256, 0, stream>>>(e16, cats, acc);
    finalize_kernel<<<1, 1, 0, stream>>>(acc, out);
}
